// MambaLayer_3195455668581
// MI455X (gfx1250) — compile-verified, matrix operand order checked by reading, not run on hardware
//
#include <hip/hip_runtime.h>
#include <math.h>

// ---------------- model dims ----------------
#define B_    2
#define L_    1024
#define DM    1024            // d_model
#define DI    2048            // d_inner
#define DS    16              // d_state
#define DTR   64              // dt_rank
#define MTOT  (B_*L_)         // 2048 rows
#define XW    128             // x_dbl width padded 96 -> 128 (zero cols 96..127)

typedef __attribute__((ext_vector_type(16))) __bf16 v16bf;
typedef __attribute__((ext_vector_type(8)))  float  v8f;
typedef __attribute__((ext_vector_type(4)))  int    v4i;
typedef unsigned short u16;

#if __has_builtin(__builtin_amdgcn_global_load_async_to_lds_b128)
#define ASYNC_LDS 1
typedef __attribute__((address_space(1))) v4i* gv4i_p;   // global int4*
typedef __attribute__((address_space(3))) v4i* lv4i_p;   // LDS int4*
#endif

__device__ __forceinline__ u16 f2bf(float x) {
  union { __bf16 b; u16 u; } cv; cv.b = (__bf16)x; return cv.u;
}

// CDNA5 16-bit A/B fragment gather (16x16x32): lane half selects K groups.
// rowbase -> 32 contiguous bf16 (one A row / one Bt row) in LDS.
// Lowers to two ds_load_b128 (offsets 0 and 32B).
__device__ __forceinline__ v16bf load_frag(const u16* rowbase, int half) {
  union { v16bf v; unsigned int u[8]; } f;
#pragma unroll
  for (int i = 0; i < 8; ++i) {
    int k = ((i < 4) ? (2 * i) : (16 + 2 * (i - 4))) + 8 * half;
    f.u[i] = *(const unsigned int*)(rowbase + k);
  }
  return f.v;
}

// ---------------- WMMA GEMM (bf16 x bf16t -> f32), fully tile-aligned ------
// A row-major [M][K]; Bt row-major [N][K] (B pre-transposed).
// M,N % 128 == 0, K % 64 == 0. Double-buffered async global->LDS pipeline.
// EPI 0: C=acc (f32)   1: C=softplus(acc+bias[col])   2: C=acc + bf16 copy Cbf
#define TM 128
#define TN 128
#define TK 32
#define SSTR 40   // ushort stride per LDS row: 32 + 8 pad (80B, 16B-aligned)

__device__ __forceinline__ void stage_tile(
    const u16* __restrict__ A, const u16* __restrict__ Bt,
    int lda, int ldbt, int m0, int n0, int k0, int tid,
    u16* sA, u16* sB)
{
  // A tile 128x32 and Bt tile 128x32: 512 16B chunks each, 2+2 per thread.
#pragma unroll
  for (int i = 0; i < 2; ++i) {
    int idx = tid + 256 * i;
    int r = idx >> 2, c = idx & 3;                 // 4 chunks per row
    const u16* gpA = A  + (size_t)(m0 + r) * lda  + k0 + 8 * c;
    const u16* gpB = Bt + (size_t)(n0 + r) * ldbt + k0 + 8 * c;
    u16* lpA = &sA[r * SSTR + 8 * c];
    u16* lpB = &sB[r * SSTR + 8 * c];
#ifdef ASYNC_LDS
    __builtin_amdgcn_global_load_async_to_lds_b128(
        (gv4i_p)(void*)gpA, (lv4i_p)(void*)lpA, 0, 0);
    __builtin_amdgcn_global_load_async_to_lds_b128(
        (gv4i_p)(void*)gpB, (lv4i_p)(void*)lpB, 0, 0);
#else
    *(uint4*)lpA = *(const uint4*)gpA;
    *(uint4*)lpB = *(const uint4*)gpB;
#endif
  }
}

__device__ __forceinline__ void compute_step(
    const u16* sAbuf, const u16* sBbuf,
    int wm, int wn, int l16, int half, v8f (&acc)[2][4])
{
  v16bf af[2], bfr[4];
#pragma unroll
  for (int tm = 0; tm < 2; ++tm)
    af[tm] = load_frag(&sAbuf[(wm * 32 + tm * 16 + l16) * SSTR], half);
#pragma unroll
  for (int tn = 0; tn < 4; ++tn)
    bfr[tn] = load_frag(&sBbuf[(wn * 64 + tn * 16 + l16) * SSTR], half);
#pragma unroll
  for (int tm = 0; tm < 2; ++tm)
#pragma unroll
    for (int tn = 0; tn < 4; ++tn)
      acc[tm][tn] = __builtin_amdgcn_wmma_f32_16x16x32_bf16(
          false, af[tm], false, bfr[tn], (short)0, acc[tm][tn], false, false);
}

template <int EPI>
__global__ __launch_bounds__(256) void wmma_gemm_bf16(
    const u16* __restrict__ A, const u16* __restrict__ Bt,
    float* __restrict__ C, u16* __restrict__ Cbf,
    int K, int lda, int ldbt, int ldc, int ldcb,
    const float* __restrict__ bias)
{
  __shared__ u16 sA [2][TM * SSTR];
  __shared__ u16 sBt[2][TN * SSTR];

  const int tid  = threadIdx.x;
  const int lane = tid & 31;
  const int wave = tid >> 5;
  const int half = lane >> 4;
  const int l16  = lane & 15;
  const int wm   = wave & 3;    // 4 waves along M, 32 rows each
  const int wn   = wave >> 2;   // 2 waves along N, 64 cols each
  const int m0   = blockIdx.y * TM;
  const int n0   = blockIdx.x * TN;

  v8f acc[2][4];
#pragma unroll
  for (int i = 0; i < 2; ++i)
#pragma unroll
    for (int j = 0; j < 4; ++j)
#pragma unroll
      for (int g = 0; g < 8; ++g) acc[i][j][g] = 0.f;

  const int nk = K / TK;        // >= 2 for all our shapes
  stage_tile(A, Bt, lda, ldbt, m0, n0, 0, tid, sA[0], sBt[0]);

  // steady state: branch-free pipelined loop over k-tiles 0..nk-2
  for (int kt = 0; kt < nk - 1; ++kt) {
    const int cur = kt & 1;
    stage_tile(A, Bt, lda, ldbt, m0, n0, (kt + 1) * TK, tid,
               sA[cur ^ 1], sBt[cur ^ 1]);
#ifdef ASYNC_LDS
    // 4 async instrs outstanding for next tile; in-order completion means
    // cnt<=4 guarantees the current tile has fully landed in LDS.
    asm volatile("s_wait_asynccnt 0x4" ::: "memory");
#endif
    __syncthreads();
    compute_step(sA[cur], sBt[cur], wm, wn, l16, half, acc);
    __syncthreads();   // protect buffer 'cur' before it is overwritten at kt+2
  }
  // tail: last tile, no further staging
#ifdef ASYNC_LDS
  asm volatile("s_wait_asynccnt 0x0" ::: "memory");
#endif
  __syncthreads();
  compute_step(sA[(nk - 1) & 1], sBt[(nk - 1) & 1], wm, wn, l16, half, acc);

  // ---- epilogue (registers only): VGPR g -> row g + 8*half, col = lane&15
#pragma unroll
  for (int tm = 0; tm < 2; ++tm) {
#pragma unroll
    for (int tn = 0; tn < 4; ++tn) {
      int rbase = m0 + wm * 32 + tm * 16 + half * 8;
      int col   = n0 + wn * 64 + tn * 16 + l16;
      float bb = (EPI == 1) ? bias[col] : 0.f;
#pragma unroll
      for (int g = 0; g < 8; ++g) {
        int row = rbase + g;
        float v = acc[tm][tn][g];
        if (EPI == 1) {
          float s = v + bb;
          v = (s > 20.f) ? s : log1pf(__expf(s));
        }
        C[(size_t)row * ldc + col] = v;
        if (EPI == 2) Cbf[(size_t)row * ldcb + col] = f2bf(v);
      }
    }
  }
}

// ---------------- f32 -> bf16 bulk convert (8 elems/thread) ----------------
__global__ __launch_bounds__(256) void cvt_bf16(
    const float* __restrict__ s, u16* __restrict__ d, int n8)
{
  int i = blockIdx.x * 256 + threadIdx.x;
  if (i >= n8) return;
  const float4* p = (const float4*)(s + (size_t)i * 8);
  float4 a = p[0], b = p[1];
  union { uint4 q; u16 u[8]; } t;
  t.u[0] = f2bf(a.x); t.u[1] = f2bf(a.y); t.u[2] = f2bf(a.z); t.u[3] = f2bf(a.w);
  t.u[4] = f2bf(b.x); t.u[5] = f2bf(b.y); t.u[6] = f2bf(b.z); t.u[7] = f2bf(b.w);
  *(uint4*)(d + (size_t)i * 8) = t.q;
}

// Weight convert + transpose: src f32 [K][srcN] -> dst bf16 [Npad][K].
// idx over Npad*K ; rows n >= srcN are zero padding. K = 1<<kshift.
__global__ __launch_bounds__(256) void cvt_bf16_t(
    const float* __restrict__ s, u16* __restrict__ d,
    int kshift, int kmask, int srcN)
{
  int idx = blockIdx.x * 256 + threadIdx.x;
  int n = idx >> kshift, k = idx & kmask;
  d[idx] = (n < srcN) ? f2bf(s[(size_t)k * srcN + n]) : (u16)0;
}

// ---------------- causal depthwise conv1d (width 4) + SiLU -----------------
__global__ __launch_bounds__(256) void conv_silu_k(
    const float* __restrict__ xz, const float* __restrict__ ck,
    const float* __restrict__ cb, float* __restrict__ uc, u16* __restrict__ ucb)
{
  int idx = blockIdx.x * 256 + threadIdx.x;     // over MTOT*DI
  int m = idx >> 11;
  int c = idx & (DI - 1);
  int b = m >> 10, t = m & (L_ - 1);
  float acc = cb[c];
#pragma unroll
  for (int w = 0; w < 4; ++w) {
    int tt = t - 3 + w;
    if (tt >= 0)
      acc = fmaf(xz[(size_t)(b * L_ + tt) * (2 * DI) + c], ck[w * DI + c], acc);
  }
  float v = acc / (1.f + __expf(-acc));         // SiLU
  uc[idx]  = v;                                 // f32 for the scan
  ucb[idx] = f2bf(v);                           // bf16 for x_dbl GEMM
}

// ---------------- selective scan: one thread per (b, d) channel ------------
// B(t)/C(t) addresses are wave-uniform -> scalar loads (constant cache, KMcnt),
// so the scan needs no LDS and no barriers at all.
__global__ __launch_bounds__(256) void mamba_scan(
    const float* __restrict__ xdbl, const float* __restrict__ dlt,
    const float* __restrict__ uc,   const float* __restrict__ xz,
    const float* __restrict__ A_log, const float* __restrict__ Dp,
    u16* __restrict__ ygb)
{
  const int tid = threadIdx.x;
  const int b = blockIdx.x >> 3;
  const int d = ((blockIdx.x & 7) << 8) + tid;

  float Areg[DS], h[DS];
#pragma unroll
  for (int n = 0; n < DS; ++n) {
    Areg[n] = -__expf(A_log[(size_t)d * DS + n]);
    h[n] = 0.f;
  }
  const float Dd = Dp[d];

  for (int t = 0; t < L_; ++t) {
    const int m = b * L_ + t;
    const float* __restrict__ bc = xdbl + (size_t)m * XW + DTR;  // uniform addr
    const float dt = dlt[(size_t)m * DI + d];
    const float u  = uc[(size_t)m * DI + d];
    const float du = dt * u;
    float y = 0.f;
#pragma unroll
    for (int n = 0; n < DS; ++n) {
      float dA = __expf(dt * Areg[n]);
      h[n] = fmaf(dA, h[n], du * bc[n]);         // h = dA*h + dBu  (B scalar-ld)
      y = fmaf(h[n], bc[DS + n], y);             // y = <h, C>      (C scalar-ld)
    }
    y = fmaf(u, Dd, y);
    const float z = xz[(size_t)m * (2 * DI) + DI + d];
    y *= z / (1.f + __expf(-z));                 // gate with SiLU(z)
    ygb[(size_t)m * DI + d] = f2bf(y);           // bf16 for output GEMM
  }
}

// ---------------- launch ----------------
extern "C" void kernel_launch(void* const* d_in, const int* in_sizes, int n_in,
                              void* d_out, int out_size, void* d_ws, size_t ws_size,
                              hipStream_t stream) {
  const float* x     = (const float*)d_in[0];
  const float* W_in  = (const float*)d_in[1];
  const float* ck    = (const float*)d_in[2];
  const float* cb    = (const float*)d_in[3];
  const float* W_x   = (const float*)d_in[4];
  const float* W_dt  = (const float*)d_in[5];
  const float* b_dt  = (const float*)d_in[6];
  const float* W_out = (const float*)d_in[7];
  const float* A_log = (const float*)d_in[8];
  const float* Dp    = (const float*)d_in[9];
  float* out = (float*)d_out;

  char* ws = (char*)d_ws;
  size_t off = 0;
  float* xz    = (float*)(ws + off); off += (size_t)MTOT * 2 * DI * 4;  // 32 MB
  float* uc    = (float*)(ws + off); off += (size_t)MTOT * DI * 4;      // 16 MB
  float* dlt   = (float*)(ws + off); off += (size_t)MTOT * DI * 4;      // 16 MB
  float* xdbl  = (float*)(ws + off); off += (size_t)MTOT * XW * 4;      //  1 MB
  u16* xb      = (u16*)(ws + off);   off += (size_t)MTOT * DM * 2;      //  4 MB
  u16* Winbt   = (u16*)(ws + off);   off += (size_t)(2 * DI) * DM * 2;  //  8 MB [4096][1024]
  u16* ucb     = (u16*)(ws + off);   off += (size_t)MTOT * DI * 2;      //  8 MB
  u16* Wxbt    = (u16*)(ws + off);   off += (size_t)XW * DI * 2;        // .5 MB [128][2048]
  u16* xdblb   = (u16*)(ws + off);   off += (size_t)MTOT * XW * 2;      // .5 MB
  u16* Wdtbt   = (u16*)(ws + off);   off += (size_t)DI * DTR * 2;       // .25MB [2048][64]
  u16* Woutbt  = (u16*)(ws + off);   off += (size_t)DM * DI * 2;        //  4 MB [1024][2048]
  u16* ygb     = (u16*)(ws + off);   off += (size_t)MTOT * DI * 2;      //  8 MB

  dim3 blk(256);
  // 0) one-time (per launch) weight conversions: bf16 + transpose [N][K]
  cvt_bf16<<<(MTOT * DM / 8) / 256, blk, 0, stream>>>(x, xb, MTOT * DM / 8);
  cvt_bf16_t<<<((2 * DI) * DM) / 256, blk, 0, stream>>>(W_in, Winbt, 10, DM - 1, 2 * DI);
  cvt_bf16_t<<<(XW * DI) / 256, blk, 0, stream>>>(W_x, Wxbt, 11, DI - 1, 96);
  cvt_bf16_t<<<(DI * DTR) / 256, blk, 0, stream>>>(W_dt, Wdtbt, 6, DTR - 1, DI);
  cvt_bf16_t<<<(DM * DI) / 256, blk, 0, stream>>>(W_out, Woutbt, 11, DI - 1, DM);

  // 1) xz = x @ W_in                       M=2048 N=4096 K=1024
  wmma_gemm_bf16<0><<<dim3((2 * DI) / TN, MTOT / TM), blk, 0, stream>>>(
      xb, Winbt, xz, nullptr, DM, DM, DM, 2 * DI, 0, nullptr);
  // 2) u = SiLU(causal depthwise conv(xz[:, :DI]))
  conv_silu_k<<<(MTOT * DI) / 256, blk, 0, stream>>>(xz, ck, cb, uc, ucb);
  // 3) x_dbl = u @ W_x (padded)            M=2048 N=128 K=2048 (+ bf16 copy)
  wmma_gemm_bf16<2><<<dim3(XW / TN, MTOT / TM), blk, 0, stream>>>(
      ucb, Wxbt, xdbl, xdblb, DI, DI, DI, XW, XW, nullptr);
  // 4) delta = softplus(x_dbl[:, :64] @ W_dt + b_dt)   M=2048 N=2048 K=64
  wmma_gemm_bf16<1><<<dim3(DI / TN, MTOT / TM), blk, 0, stream>>>(
      xdblb, Wdtbt, dlt, nullptr, DTR, XW, DTR, DI, 0, b_dt);
  // 5) selective scan + D skip + SiLU(z) gate -> bf16 y (no barriers)
  mamba_scan<<<B_ * (DI / 256), blk, 0, stream>>>(xdbl, dlt, uc, xz, A_log, Dp, ygb);
  // 6) out = y @ W_out                     M=2048 N=1024 K=2048
  wmma_gemm_bf16<0><<<dim3(DM / TN, MTOT / TM), blk, 0, stream>>>(
      ygb, Woutbt, out, nullptr, DI, DI, DI, DM, 0, nullptr);
}
